// moe_gating_73014444032628
// MI455X (gfx1250) — matrix operand order repaired and checked by reading, not compiled, not run on hardware
//
#include <hip/hip_runtime.h>

// ---------------------------------------------------------------------------
// MoE (gate + top-3 + 2-layer GELU FFN over 6 experts) for MI455X / gfx1250.
//
// B=2, C=256, N=32768 tokens/batch, E=6, HID=1024, TOPK=3. Compute-bound
// (~412 GFLOP of GEMM, weights L2-resident) -> v_wmma_f32_16x16x32_bf16.
// Transposed-tile formulation (no transposes needed anywhere):
//   GEMM1: W1(16h x 32c) * X^T(32c x 16t)  -> hdn   (h-major x tokens)
//   GEMM2: W2(16c x 32h) * hdn(32h x 16t)  -> out   (c-major == [B,C,N])
// Top-k mask folded into GEMM2's B operand (column scaling of hdn) so one
// persistent WMMA accumulator sums all 6 experts; output written once.
//
// v3: each wave now computes TWO M-tiles per B fragment (16 waves = 8 m-pair
// groups x 2 token halves) -> LDS B-fragment bandwidth per WMMA halved (one
// ds_load_b128 pair feeds two WMMAs), accumulators stay at 64 VGPRs so the
// kernel remains under the 256-VGPR MSB boundary. Weights pre-packed to bf16
// per-lane A-fragment layout (1024B contiguous per wave-fragment, zero VALU).
// ---------------------------------------------------------------------------

typedef __attribute__((ext_vector_type(16))) __bf16 bf16x16;
typedef __attribute__((ext_vector_type(8)))  float  f32x8;

static constexpr int kC    = 256;
static constexpr int kN    = 32768;          // tokens per batch
static constexpr int kNT   = 2 * kN;         // 65536 total tokens
static constexpr int kE    = 6;
static constexpr int kHID  = 1024;
static constexpr int kTB   = 128;            // tokens per block
static constexpr int kXST  = kTB + 8;        // LDS row stride (halves): 272B -> no bank conflicts

__device__ __forceinline__ __bf16 f32_to_bf16(float f) {
  unsigned int u = __builtin_bit_cast(unsigned int, f);
  unsigned int r = (u + 0x7FFFu + ((u >> 16) & 1u)) >> 16;  // round-to-nearest-even
  return __builtin_bit_cast(__bf16, (unsigned short)r);
}

__device__ __forceinline__ f32x8 zero8() {
  f32x8 z;
#pragma unroll
  for (int i = 0; i < 8; ++i) z[i] = 0.0f;
  return z;
}

struct U32x8 { uint4 lo; uint4 hi; };

// B fragment (32k x 16n, bf16) from LDS: lane l holds column n=(l&15),
// halves[0..7] = k kb+0..7, halves[8..15] = k kb+16..23, kb=8*(l>>4).
__device__ __forceinline__ bf16x16 load_b_lds(const __bf16* buf, int koff, int toff, int lane) {
  const int n  = toff + (lane & 15);
  const int kb = koff + ((lane >> 4) << 3);
  bf16x16 v;
#pragma unroll
  for (int i = 0; i < 8; ++i) {
    v[i]     = buf[(kb + i) * kXST + n];
    v[i + 8] = buf[(kb + 16 + i) * kXST + n];
  }
  return v;
}

// A fragment from pre-packed bf16 weights: fragment = 512 contiguous halves,
// lane l owns halves [l*16, l*16+16) -> wave reads 1024 contiguous bytes.
__device__ __forceinline__ bf16x16 load_a_packed(const __bf16* frag, int lane) {
  const uint4* p = (const uint4*)frag + lane * 2;
  U32x8 s;
  s.lo = p[0];
  s.hi = p[1];
  return __builtin_bit_cast(bf16x16, s);
}

__device__ __forceinline__ f32x8 wmma_bf16(bf16x16 a, bf16x16 b, f32x8 c) {
  return __builtin_amdgcn_wmma_f32_16x16x32_bf16(false, a, false, b, (short)0, c, false, false);
}

// tanh-form GELU: __expf is a TRANS op that co-executes with XDL WMMA.
__device__ __forceinline__ float gelu_f(float v) {
  float u  = 0.7978845608f * (v + 0.044715f * v * v * v);
  float e  = __expf(2.0f * u);
  float th = (e - 1.0f) / (e + 1.0f);
  return 0.5f * v * (1.0f + th);
}

// ---------------------------------------------------------------------------
// Pre-pack: row-major f32 weight set [E][R][K] -> bf16 A-fragments.
// Fragment index = (e*(R/16) + mtile)*(K/32) + kstep; within a fragment,
// lane l holds: halves[0..7] = row (mtile*16 + (l&15)), cols kstep*32+kb+0..7
//               halves[8..15] = same row, cols kstep*32+kb+16..23, kb=(l>>4)*8.
// One thread per (fragment, lane): writes 32 contiguous bytes (coalesced).
// ---------------------------------------------------------------------------
__global__ __launch_bounds__(256) void pack_a_kernel(const float* __restrict__ src,
                                                     __bf16* __restrict__ dst,
                                                     int R, int K) {
  const int tid  = blockIdx.x * 256 + threadIdx.x;
  const int lane = tid & 31;
  const int frag = tid >> 5;
  const int kt   = K >> 5;                  // ksteps per row
  const int mt   = R >> 4;                  // mtiles per matrix
  const int ks   = frag % kt;
  const int m    = (frag / kt) % mt;
  const int e    = frag / (kt * mt);
  if (e >= kE) return;

  const float* p = src + ((size_t)e * R + m * 16 + (lane & 15)) * K
                       + ks * 32 + ((lane >> 4) << 3);
  __bf16 h[16];
#pragma unroll
  for (int i = 0; i < 8; ++i) {
    h[i]     = f32_to_bf16(p[i]);
    h[i + 8] = f32_to_bf16(p[i + 16]);
  }
  U32x8 s = __builtin_bit_cast(U32x8, *(const bf16x16*)h);
  uint4* d = (uint4*)(dst + (size_t)tid * 16);
  d[0] = s.lo;
  d[1] = s.hi;
}

// ---------------------------------------------------------------------------
// Gating: 6 logits, *attention, softmax over E, top-3 -> mask[token][8] in ws.
// ---------------------------------------------------------------------------
__global__ __launch_bounds__(256) void moe_gate_kernel(
    const float* __restrict__ x, const float* __restrict__ attn,
    const float* __restrict__ gw, const float* __restrict__ gb,
    float* __restrict__ mask) {
  __shared__ float sgw[kE * kC];
  for (int i = threadIdx.x; i < kE * kC; i += 256) sgw[i] = gw[i];
  __syncthreads();

  const int g    = blockIdx.x * 256 + threadIdx.x;
  const int bidx = g / kN;
  const int n    = g - bidx * kN;
  const float* xb = x + (size_t)bidx * kC * kN + n;

  float acc[kE];
#pragma unroll
  for (int e = 0; e < kE; ++e) acc[e] = gb[e];
  for (int c = 0; c < kC; ++c) {
    float xv = xb[(size_t)c * kN];
#pragma unroll
    for (int e = 0; e < kE; ++e) acc[e] += xv * sgw[e * kC + c];
  }
  const float aw = attn[g];
#pragma unroll
  for (int e = 0; e < kE; ++e) acc[e] *= aw;

  float mx = acc[0];
#pragma unroll
  for (int e = 1; e < kE; ++e) mx = fmaxf(mx, acc[e]);
  float p[kE], s = 0.0f;
#pragma unroll
  for (int e = 0; e < kE; ++e) { p[e] = __expf(acc[e] - mx); s += p[e]; }
  const float inv = 1.0f / s;
#pragma unroll
  for (int e = 0; e < kE; ++e) p[e] *= inv;

  unsigned sel = 0;
#pragma unroll
  for (int it = 0; it < 3; ++it) {
    int best = 0; float bv = -1.0f;
#pragma unroll
    for (int e = 0; e < kE; ++e)
      if (!((sel >> e) & 1u) && p[e] > bv) { bv = p[e]; best = e; }
    sel |= 1u << best;
  }
#pragma unroll
  for (int e = 0; e < 8; ++e)
    mask[(size_t)g * 8 + e] = (e < kE && ((sel >> e) & 1u)) ? p[e] : 0.0f;
}

// ---------------------------------------------------------------------------
// Fused dual-GEMM FFN. Block = 128 tokens, 16 waves (512 threads).
// Wave decomposition: wave = th*8 + mp, th = token half (4 of the 8 token
// tiles), mp = M-pair group. Each wave owns M-tiles {2mp, 2mp+1} for both
// GEMMs -> one LDS B fragment feeds two WMMAs. Weights read once per block.
// ---------------------------------------------------------------------------
__global__ __launch_bounds__(512) void moe_ffn_kernel(
    const float* __restrict__ x,
    const __bf16* __restrict__ w1p,   // packed: ((e*64+mt)*8+k)*512 halves
    const float* __restrict__ b1,
    const __bf16* __restrict__ w2p,   // packed: ((e*16+ct)*32+ks)*512 halves
    const float* __restrict__ b2,
    const float* __restrict__ mask, float* __restrict__ out) {
  extern __shared__ __bf16 smem[];
  __bf16* xs = smem;                 // [256c][kXST] bf16 x tile   (B op, GEMM1)
  __bf16* hs = smem + kC * kXST;     // [256h][kXST] bf16 hdn chunk(B op, GEMM2)

  const int lane = threadIdx.x & 31;
  const int wave = threadIdx.x >> 5;   // 0..15
  const int mp   = wave & 7;           // M-pair group (2 mtiles each)
  const int th   = wave >> 3;          // token half: tt in [th*4, th*4+4)
  const int tt0  = th * 4;
  const int tloc = lane & 15;
  const int csel = lane >> 4;
  const int gt0  = blockIdx.x * kTB;
  const int bidx = gt0 / kN;
  const int n0   = gt0 - bidx * kN;
  const float* xb = x + (size_t)bidx * kC * kN + n0;

  // Stage x tile: f32 [C][128] -> bf16 LDS, coalesced along tokens.
  for (int idx = threadIdx.x; idx < kC * kTB; idx += 512) {
    int c = idx >> 7, t = idx & (kTB - 1);
    xs[c * kXST + t] = f32_to_bf16(xb[(size_t)c * kN + t]);
  }
  __syncthreads();

  f32x8 acc[2][4];                   // out tile: c rows [mp*32,+32) x 4 token tiles
#pragma unroll
  for (int mi = 0; mi < 2; ++mi)
#pragma unroll
    for (int j = 0; j < 4; ++j) acc[mi][j] = zero8();

  for (int e = 0; e < kE; ++e) {
    float mk[4];
#pragma unroll
    for (int j = 0; j < 4; ++j)
      mk[j] = mask[(size_t)(gt0 + (tt0 + j) * 16 + tloc) * 8 + e];

    const __bf16* w1e = w1p + (size_t)e * 64 * 8 * 512;    // 64 mtiles x 8 ksteps
    const __bf16* w2e = w2p + (size_t)e * 16 * 32 * 512;   // 16 ctiles x 32 ksteps
    const float*  b1e = b1 + e * kHID;

    for (int hc = 0; hc < 4; ++hc) {           // HID in chunks of 256
      // --- GEMM1: wave's h mtiles {hc*16+2mp, +1}, K = C = 256 ---
      {
        const int mt = hc * 16 + mp * 2;        // global h mtile
        f32x8 hacc[2][4];
#pragma unroll
        for (int mi = 0; mi < 2; ++mi)
#pragma unroll
          for (int j = 0; j < 4; ++j) hacc[mi][j] = zero8();
        for (int k = 0; k < 8; ++k) {
          bf16x16 a0 = load_a_packed(w1e + ((size_t)(mt + 0) * 8 + k) * 512, lane);
          bf16x16 a1 = load_a_packed(w1e + ((size_t)(mt + 1) * 8 + k) * 512, lane);
#pragma unroll
          for (int j = 0; j < 4; ++j) {
            bf16x16 bfr = load_b_lds(xs, k * 32, (tt0 + j) * 16, lane);
            hacc[0][j] = wmma_bf16(a0, bfr, hacc[0][j]);   // one B feeds two WMMAs
            hacc[1][j] = wmma_bf16(a1, bfr, hacc[1][j]);
          }
        }
        // Epilogue: +b1, GELU, *mask (per-token column scale), -> LDS bf16.
#pragma unroll
        for (int mi = 0; mi < 2; ++mi) {
          const int hl0 = (mp * 2 + mi) * 16;   // row within chunk
#pragma unroll
          for (int r = 0; r < 8; ++r) {
            const int hl   = hl0 + r + (csel << 3);
            const float bi = b1e[hc * 256 + hl];
#pragma unroll
            for (int j = 0; j < 4; ++j) {
              float v = gelu_f(hacc[mi][j][r] + bi) * mk[j];
              hs[hl * kXST + (tt0 + j) * 16 + tloc] = f32_to_bf16(v);
            }
          }
        }
      }
      __syncthreads();
      // --- GEMM2: wave's c mtiles {2mp, 2mp+1}, K = chunk's 256 h values ---
      {
        const int ct = mp * 2;
        for (int k = 0; k < 8; ++k) {
          const int ks = hc * 8 + k;            // global kstep 0..31
          bf16x16 a0 = load_a_packed(w2e + ((size_t)(ct + 0) * 32 + ks) * 512, lane);
          bf16x16 a1 = load_a_packed(w2e + ((size_t)(ct + 1) * 32 + ks) * 512, lane);
#pragma unroll
          for (int j = 0; j < 4; ++j) {
            bf16x16 bfr = load_b_lds(hs, k * 32, (tt0 + j) * 16, lane);
            acc[0][j] = wmma_bf16(a0, bfr, acc[0][j]);
            acc[1][j] = wmma_bf16(a1, bfr, acc[1][j]);
          }
        }
      }
      __syncthreads();
    }
  }

  // Final: add sum_e mask[t][e]*b2[e][c] and store out[B,C,N] (written once).
  float* outb = out + (size_t)bidx * kC * kN + n0;
#pragma unroll
  for (int j = 0; j < 4; ++j) {
    const int t = (tt0 + j) * 16 + tloc;
    float m6[kE];
#pragma unroll
    for (int e = 0; e < kE; ++e) m6[e] = mask[(size_t)(gt0 + t) * 8 + e];
#pragma unroll
    for (int mi = 0; mi < 2; ++mi) {
#pragma unroll
      for (int r = 0; r < 8; ++r) {
        const int c = (mp * 2 + mi) * 16 + r + (csel << 3);
        float bsum = 0.0f;
#pragma unroll
        for (int e = 0; e < kE; ++e) bsum += m6[e] * b2[e * kC + c];
        outb[(size_t)c * kN + t] = acc[mi][j][r] + bsum;
      }
    }
  }
}

// ---------------------------------------------------------------------------
extern "C" void kernel_launch(void* const* d_in, const int* in_sizes, int n_in,
                              void* d_out, int out_size, void* d_ws, size_t ws_size,
                              hipStream_t stream) {
  (void)in_sizes; (void)n_in; (void)out_size; (void)ws_size;
  const float* x    = (const float*)d_in[0];
  const float* attn = (const float*)d_in[1];
  const float* gw   = (const float*)d_in[2];
  const float* gb   = (const float*)d_in[3];
  const float* w1   = (const float*)d_in[4];
  const float* b1   = (const float*)d_in[5];
  const float* w2   = (const float*)d_in[6];
  const float* b2   = (const float*)d_in[7];
  float* out = (float*)d_out;

  // Workspace layout: mask (kNT*8 f32 = 2MB) | w1 packed (3MB) | w2 packed (3MB)
  float*  mask = (float*)d_ws;
  __bf16* w1p  = (__bf16*)((char*)d_ws + (size_t)kNT * 8 * sizeof(float));
  __bf16* w2p  = w1p + (size_t)kE * kHID * kC;

  // One-time per launch: pack weights to bf16 fragment layout (coalesced writes).
  pack_a_kernel<<<(kE * (kHID / 16) * (kC / 32) * 32) / 256, 256, 0, stream>>>(w1, w1p, kHID, kC);
  pack_a_kernel<<<(kE * (kC / 16) * (kHID / 32) * 32) / 256, 256, 0, stream>>>(w2, w2p, kC, kHID);

  moe_gate_kernel<<<kNT / 256, 256, 0, stream>>>(x, attn, gw, gb, mask);

  const size_t ldsBytes = (size_t)2 * kC * kXST * sizeof(__bf16);  // ~139 KB (320KB/WGP on CDNA5)
  moe_ffn_kernel<<<kNT / kTB, 512, ldsBytes, stream>>>(x, w1p, b1, w2p, b2, mask, out);
}
